// EdgePropertyPredictor_30374008717369
// MI455X (gfx1250) — compile-verified, matrix operand order checked by reading, not run on hardware
//
#include <hip/hip_runtime.h>
#include <hip/hip_bf16.h>
#include <stdint.h>

// ---------------------------------------------------------------------------
// EdgePropertyPredictor for MI455X (gfx1250, wave32, WMMA).
// Pipeline: [pack weights f32->bf16 fragment-major] -> [feature build]
//           -> 3x fused (bf16 WMMA GEMM + bias + ELU + residual)
//           -> final (bf16 WMMA GEMM + bias) -> fp32 out.
// GEMM: block tile 128x128, wave tile 32x64; A double-buffered in LDS,
//       B fragments loaded straight from L2 in pre-packed fragment layout,
//       register-double-buffered one K-step ahead (loop unrolled x2).
// ---------------------------------------------------------------------------

typedef __attribute__((ext_vector_type(16))) __bf16 v16bf;
typedef __attribute__((ext_vector_type(8)))  float  v8f;
typedef __attribute__((ext_vector_type(4)))  unsigned int v4u;

union FragB16 { v16bf v; v4u u4[2]; unsigned int u[8]; };
union Pack8   { v4u q; unsigned short s[8]; };

#define NE    32768   // edges (GEMM M)
#define NH    256     // hidden features per node
#define ND    1280    // dense dim
#define NOUT  512     // output dim
#define NKSUB 24      // padded subgraph width

#define BM 128        // block tile M
#define BN 128        // block tile N
#define LDP 40        // padded LDS row length in shorts (80B: 16B-aligned rows)

__device__ __forceinline__ unsigned short f2bf(float f) {
  // round-to-nearest-even f32 -> bf16
  unsigned int u = __float_as_uint(f);
  u += 0x7FFFu + ((u >> 16) & 1u);
  return (unsigned short)(u >> 16);
}
__device__ __forceinline__ float bf2f(unsigned short s) {
  return __uint_as_float(((unsigned int)s) << 16);
}

// ---------------------------------------------------------------------------
// Weight pack: f32 [K,N] row-major -> bf16 fragment-major.
// Block (kt, g) = 32x16 sub-matrix stored as 32 lanes x 16 shorts (32B/lane),
// lane l: col n = g*16 + (l&15), kh = (l>>4)*8; short j: ISA 7.12.2 K-pairs.
// ---------------------------------------------------------------------------
__global__ __launch_bounds__(256) void pack_weights_kernel(
    const float* __restrict__ W, unsigned short* __restrict__ WP, int K, int N) {
  const int NG = N >> 4;
  const int total = K * N;
  for (int id = blockIdx.x * blockDim.x + threadIdx.x; id < total;
       id += gridDim.x * blockDim.x) {
    int j    = id & 15;
    int lane = (id >> 4) & 31;
    int blk  = id >> 9;
    int g    = blk % NG;
    int kt   = blk / NG;
    int n    = g * 16 + (lane & 15);
    int kh   = (lane >> 4) * 8;
    int jd   = j >> 1;
    int kbase = (jd < 4) ? (kh + 2 * jd) : (16 + kh + 2 * (jd - 4));
    int kk   = (kt << 5) + kbase + (j & 1);
    WP[id] = f2bf(W[(size_t)kk * N + n]);
  }
}

// ---------------------------------------------------------------------------
// Feature build: one block (256 thr) per edge.
// Mask semantics match the reference: flat = idx + off, masked only when
// flat == -1 (padding rows in graphs with off>0 DO contribute X[off-1]).
// ---------------------------------------------------------------------------
__global__ __launch_bounds__(256) void build_feats_kernel(
    const float* __restrict__ X,
    const int*   __restrict__ edge_index,   // [2, NE]
    const int*   __restrict__ batch_vec,    // [N]
    const int*   __restrict__ ptrv,         // [B+1]
    const int*   __restrict__ sgl,          // [NE, NKSUB]
    const int*   __restrict__ sgr,          // [NE, NKSUB]
    const float* __restrict__ elem,         // [NE, 24]
    const float* __restrict__ emb,          // [NE, 200]
    const float* __restrict__ stat,         // [NE, 32]
    unsigned short* __restrict__ feats)     // [NE, ND] bf16
{
  const int e = blockIdx.x;
  const int t = threadIdx.x;
  const int src = edge_index[e];
  const int dst = edge_index[NE + e];
  const int off = ptrv[batch_vec[src]];
  unsigned short* fe = feats + (size_t)e * ND;

  fe[t]      = f2bf(X[(size_t)src * NH + t]);
  fe[NH + t] = f2bf(X[(size_t)dst * NH + t]);

  float accL = 0.f, accR = 0.f;
  int cL = 0, cR = 0;
#pragma unroll 4
  for (int k = 0; k < NKSUB; ++k) {
    int fl = sgl[(size_t)e * NKSUB + k] + off;
    if (fl != -1) { accL += X[(size_t)fl * NH + t]; ++cL; }
    int fr = sgr[(size_t)e * NKSUB + k] + off;
    if (fr != -1) { accR += X[(size_t)fr * NH + t]; ++cR; }
  }
  fe[2 * NH + t] = f2bf(cL > 0 ? accL / (float)cL : 0.f);
  fe[3 * NH + t] = f2bf(cR > 0 ? accR / (float)cR : 0.f);

  if (t < 24)  fe[1024 + t] = f2bf(elem[(size_t)e * 24 + t]);
  if (t < 200) fe[1048 + t] = f2bf(emb[(size_t)e * 200 + t]);
  if (t < 32)  fe[1248 + t] = f2bf(stat[(size_t)e * 32 + t]);
}

// ---------------------------------------------------------------------------
// One K-step of the GEMM pipeline (force-inlined; called with compile-time
// buffer parity so fbC/fbN ping-pong without register copies).
// ---------------------------------------------------------------------------
__device__ __forceinline__ void kstep(
    int ktI, int KTiles, int curbuf,
    FragB16 (&fbC)[4], FragB16 (&fbN)[4], v8f (&acc)[2][4],
    unsigned short (&As)[2][BM][LDP],
    const unsigned short* __restrict__ A,
    const unsigned short* __restrict__ WP,
    int K, int NG, size_t rowBase, int gBase,
    int ar0, int ac0, int ar1, int ac1,
    int m0, int lrow, int kh, int lane)
{
  const bool have = (ktI + 1 < KTiles);
  Pack8 aQ0, aQ1;
  if (have) {
    const int k0 = (ktI + 1) << 5;
    aQ0.q = *(const v4u*)(A + (rowBase + ar0) * K + k0 + ac0);
    aQ1.q = *(const v4u*)(A + (rowBase + ar1) * K + k0 + ac1);
    const unsigned short* wp =
        WP + ((size_t)(ktI + 1) * NG + gBase) * 512 + (size_t)lane * 16;
#pragma unroll
    for (int ni = 0; ni < 4; ++ni) {
      fbN[ni].u4[0] = *(const v4u*)(wp + ni * 512);
      fbN[ni].u4[1] = *(const v4u*)(wp + ni * 512 + 8);
    }
  }

  FragB16 fa[2];
#pragma unroll
  for (int mi = 0; mi < 2; ++mi) {
    const unsigned short* ap = &As[curbuf][m0 + mi * 16 + lrow][0];
    fa[mi].u4[0] = *(const v4u*)(ap + kh);
    fa[mi].u4[1] = *(const v4u*)(ap + 16 + kh);
  }

#pragma unroll
  for (int mi = 0; mi < 2; ++mi)
#pragma unroll
    for (int ni = 0; ni < 4; ++ni)
      acc[mi][ni] = __builtin_amdgcn_wmma_f32_16x16x32_bf16(
          false, fa[mi].v, false, fbC[ni].v, (short)0, acc[mi][ni],
          false, false);

  if (have) {
    *(v4u*)(&As[curbuf ^ 1][ar0][ac0]) = aQ0.q;
    *(v4u*)(&As[curbuf ^ 1][ar1][ac1]) = aQ1.q;
  }
  __syncthreads();
}

// ---------------------------------------------------------------------------
// Fused GEMM: out = epilogue(A[M,K](bf16) @ W(packed bf16) + bias)
//   mode 0: z = elu(z + bias) + resid   (hidden layer, bf16 out)
//   mode 1: z = z + bias                (output layer, fp32 out)
// ---------------------------------------------------------------------------
__global__ __launch_bounds__(256) void gemm_bf16_kernel(
    const unsigned short* __restrict__ A,      // [M,K] bf16
    const unsigned short* __restrict__ WP,     // packed fragment-major weights
    const float*          __restrict__ bias,   // [N]
    const unsigned short* __restrict__ resid,  // [M,N] bf16 (mode 0)
    unsigned short*       __restrict__ outB,   // [M,N] bf16 (mode 0)
    float*                __restrict__ outF,   // [M,N] f32  (mode 1)
    int M, int K, int N, int mode)
{
  __shared__ unsigned short As[2][BM][LDP];

  const int tid  = threadIdx.x;
  const int lane = tid & 31;
  const int wave = tid >> 5;
  const size_t rowBase = (size_t)blockIdx.x * BM;
  const int KTiles = K >> 5;
  const int NG = N >> 4;

  // A staging geometry: 2 b128 runs/thread over the 128x32 tile
  const int ar0 = tid >> 2,         ac0 = (tid & 3) * 8;
  const int ar1 = (tid + 256) >> 2, ac1 = ((tid + 256) & 3) * 8;

  const int m0 = (wave & 3) * 32;
  const int n0 = (wave >> 2) * 64;
  const int lrow = lane & 15;
  const int kh = (lane >> 4) * 8;
  const int gBase = blockIdx.y * (BN / 16) + (wave >> 2) * 4;

  // Preload B fragments for kt=0 and stage A tile 0
  FragB16 fbA[4], fbB[4];
  {
    const unsigned short* wp = WP + (size_t)gBase * 512 + (size_t)lane * 16;
#pragma unroll
    for (int ni = 0; ni < 4; ++ni) {
      fbA[ni].u4[0] = *(const v4u*)(wp + ni * 512);
      fbA[ni].u4[1] = *(const v4u*)(wp + ni * 512 + 8);
    }
    *(v4u*)(&As[0][ar0][ac0]) = *(const v4u*)(A + (rowBase + ar0) * K + ac0);
    *(v4u*)(&As[0][ar1][ac1]) = *(const v4u*)(A + (rowBase + ar1) * K + ac1);
  }
  __syncthreads();

  v8f acc[2][4];
#pragma unroll
  for (int mi = 0; mi < 2; ++mi)
#pragma unroll
    for (int ni = 0; ni < 4; ++ni)
      acc[mi][ni] = (v8f){0.f, 0.f, 0.f, 0.f, 0.f, 0.f, 0.f, 0.f};

  int kt = 0;
  for (; kt + 2 <= KTiles; kt += 2) {
    kstep(kt,     KTiles, 0, fbA, fbB, acc, As, A, WP, K, NG, rowBase, gBase,
          ar0, ac0, ar1, ac1, m0, lrow, kh, lane);
    kstep(kt + 1, KTiles, 1, fbB, fbA, acc, As, A, WP, K, NG, rowBase, gBase,
          ar0, ac0, ar1, ac1, m0, lrow, kh, lane);
  }
  if (kt < KTiles) {
    kstep(kt, KTiles, 0, fbA, fbB, acc, As, A, WP, K, NG, rowBase, gBase,
          ar0, ac0, ar1, ac1, m0, lrow, kh, lane);
  }

  // Epilogue (C/D layout: lane<16 -> rows 0..7, lane>=16 -> rows 8..15)
  const size_t colBase = (size_t)blockIdx.y * BN;
  const int mg = (lane >> 4) * 8;
#pragma unroll
  for (int mi = 0; mi < 2; ++mi) {
    const size_t mbase = rowBase + m0 + mi * 16 + mg;
#pragma unroll
    for (int ni = 0; ni < 4; ++ni) {
      const size_t ncol = colBase + n0 + ni * 16 + lrow;
      const float bv = bias[ncol];
#pragma unroll
      for (int r = 0; r < 8; ++r) {
        const size_t m = mbase + r;
        float z = acc[mi][ni][r] + bv;
        if (mode == 0) {
          z = (z > 0.f ? z : __expf(z) - 1.f) + bf2f(resid[m * N + ncol]);
          outB[m * N + ncol] = f2bf(z);
        } else {
          outF[m * N + ncol] = z;
        }
      }
    }
  }
}

// ---------------------------------------------------------------------------
// Launch
// ---------------------------------------------------------------------------
extern "C" void kernel_launch(void* const* d_in, const int* in_sizes, int n_in,
                              void* d_out, int out_size, void* d_ws, size_t ws_size,
                              hipStream_t stream) {
  (void)in_sizes; (void)n_in; (void)out_size; (void)ws_size;

  const float* X          = (const float*)d_in[0];
  const int*   edge_index = (const int*)d_in[1];
  const int*   batch_vec  = (const int*)d_in[2];
  const int*   ptrv       = (const int*)d_in[3];
  const int*   sgl        = (const int*)d_in[4];
  const int*   sgr        = (const int*)d_in[5];
  const float* elem       = (const float*)d_in[6];
  const float* emb        = (const float*)d_in[7];
  const float* stat       = (const float*)d_in[8];
  const float* Wd         = (const float*)d_in[9];
  const float* bd         = (const float*)d_in[10];
  const float* Wo         = (const float*)d_in[11];
  const float* bo         = (const float*)d_in[12];

  // Workspace layout (bytes): act0 | act1 | WdPacked | WoPacked (~179 MB)
  const size_t actBytes = (size_t)NE * ND * sizeof(unsigned short);
  unsigned short* act0 = (unsigned short*)d_ws;
  unsigned short* act1 = (unsigned short*)((char*)d_ws + actBytes);
  unsigned short* WdP  = (unsigned short*)((char*)d_ws + 2 * actBytes);
  unsigned short* WoP  = WdP + (size_t)3 * ND * ND;

  // 1) weights -> bf16, shuffled to WMMA fragment-major layout
  for (int l = 0; l < 3; ++l)
    pack_weights_kernel<<<2048, 256, 0, stream>>>(
        Wd + (size_t)l * ND * ND, WdP + (size_t)l * ND * ND, ND, ND);
  pack_weights_kernel<<<1024, 256, 0, stream>>>(Wo, WoP, ND, NOUT);

  // 2) feature construction -> act0 (bf16 [NE, ND])
  build_feats_kernel<<<NE, 256, 0, stream>>>(X, edge_index, batch_vec, ptrv,
                                             sgl, sgr, elem, emb, stat, act0);

  // 3) three residual dense layers (ping-pong act buffers)
  dim3 gHid(NE / BM, ND / BN);  // (256, 10)
  gemm_bf16_kernel<<<gHid, 256, 0, stream>>>(act0, WdP,                       bd,
                                             act0, act1, nullptr, NE, ND, ND, 0);
  gemm_bf16_kernel<<<gHid, 256, 0, stream>>>(act1, WdP + (size_t)ND * ND,     bd + ND,
                                             act1, act0, nullptr, NE, ND, ND, 0);
  gemm_bf16_kernel<<<gHid, 256, 0, stream>>>(act0, WdP + (size_t)2 * ND * ND, bd + 2 * ND,
                                             act0, act1, nullptr, NE, ND, ND, 0);

  // 4) output projection -> d_out (fp32 [NE, NOUT])
  dim3 gOut(NE / BM, NOUT / BN);  // (256, 4)
  gemm_bf16_kernel<<<gOut, 256, 0, stream>>>(act1, WoP, bo,
                                             nullptr, nullptr, (float*)d_out,
                                             NE, ND, NOUT, 1);
}
